// ImprovedSpectralFilter_85048942396195
// MI455X (gfx1250) — compile-verified, matrix operand order checked by reading, not run on hardware
//
#include <hip/hip_runtime.h>
#include <hip/hip_bf16.h>

// ---------------------------------------------------------------------------
// MI455X (gfx1250) implementation.
//   B=64, NPG=1024, K=32, DIN=DOUT=512, EMB=32, HEADS=4, HD=8
// Pipeline:
//   k1: per-graph eigen-encoder + MHSA + filter MLP  -> filters[B*K]   (scalar)
//   k2: x_freq[b] = V_b^T X_b (WMMA bf16, M=32 N=512 K=1024); each wave
//       computes BOTH 16-row M-strips so every x fragment is loaded once.
//   kc: Wp f32 -> bf16                                                 -> ws
//   k3: fused  S = (V.diag(f)) @ x_freq  ->  out = LN(S@Wp^T + bp)
//       128-row tiles, S and staged output in LDS (bf16), WMMA bf16,
//       Wp stream prefetched 2 K-steps ahead (global_prefetch_b8),
//       LayerNorm stats via 16-lane shfl_xor butterflies.
// ---------------------------------------------------------------------------

typedef __attribute__((ext_vector_type(16))) __bf16 v16bf;
typedef __attribute__((ext_vector_type(8)))  float  v8f;

#define NB    64
#define NPG   1024
#define KEIG  32
#define DIN   512
#define DOUT  512

// --------------------------- kernel 1: filters ------------------------------
// one block per graph, 32 threads (one wave32), thread t = eigen token t
__global__ __launch_bounds__(32) void filters_kernel(
    const float* __restrict__ eigenvalues, const unsigned char* __restrict__ eig_mask,
    const float* __restrict__ W1, const float* __restrict__ b1,
    const float* __restrict__ g1, const float* __restrict__ be1,
    const float* __restrict__ W2, const float* __restrict__ b2,
    const float* __restrict__ g2, const float* __restrict__ be2,
    const float* __restrict__ Wqkv, const float* __restrict__ bqkv,
    const float* __restrict__ Wo, const float* __restrict__ bo,
    const float* __restrict__ Wf1, const float* __restrict__ bf1,
    const float* __restrict__ Wf2, const float* __restrict__ bf2,
    float* __restrict__ filters)
{
    __shared__ float ksh[KEIG][32];
    __shared__ float vsh[KEIG][32];
    __shared__ unsigned char padsh[KEIG];

    const int b = blockIdx.x;
    const int t = threadIdx.x;            // token 0..31

    const float ev = eigenvalues[b * KEIG + t];
    padsh[t] = eig_mask[b * KEIG + t] ? 0 : 1;   // 1 = masked out

    float h[32];
    #pragma unroll
    for (int j = 0; j < 32; ++j) h[j] = ev * W1[j] + b1[j];

    // LN + relu
    {
        float m = 0.f;
        #pragma unroll
        for (int j = 0; j < 32; ++j) m += h[j];
        m *= (1.f / 32.f);
        float v = 0.f;
        #pragma unroll
        for (int j = 0; j < 32; ++j) { float d = h[j] - m; v += d * d; }
        float rs = rsqrtf(v * (1.f / 32.f) + 1e-5f);
        #pragma unroll
        for (int j = 0; j < 32; ++j) {
            float x = (h[j] - m) * rs * g1[j] + be1[j];
            h[j] = x > 0.f ? x : 0.f;
        }
    }
    // h2 = LN(h @ W2^T + b2)
    float h2[32];
    #pragma unroll
    for (int j = 0; j < 32; ++j) {
        float s = b2[j];
        #pragma unroll
        for (int i = 0; i < 32; ++i) s += h[i] * W2[j * 32 + i];
        h2[j] = s;
    }
    {
        float m = 0.f;
        #pragma unroll
        for (int j = 0; j < 32; ++j) m += h2[j];
        m *= (1.f / 32.f);
        float v = 0.f;
        #pragma unroll
        for (int j = 0; j < 32; ++j) { float d = h2[j] - m; v += d * d; }
        float rs = rsqrtf(v * (1.f / 32.f) + 1e-5f);
        #pragma unroll
        for (int j = 0; j < 32; ++j) h2[j] = (h2[j] - m) * rs * g2[j] + be2[j];
    }
    // qkv
    float q[32];
    #pragma unroll
    for (int j = 0; j < 32; ++j) {
        float sq = bqkv[j], sk = bqkv[32 + j], sv = bqkv[64 + j];
        #pragma unroll
        for (int i = 0; i < 32; ++i) {
            sq += h2[i] * Wqkv[j * 32 + i];
            sk += h2[i] * Wqkv[(32 + j) * 32 + i];
            sv += h2[i] * Wqkv[(64 + j) * 32 + i];
        }
        q[j] = sq; ksh[t][j] = sk; vsh[t][j] = sv;
    }
    __syncthreads();

    // attention: 4 heads x hd 8 over 32 tokens
    float ctx[32];
    const float scale = 0.35355339059327373f;  // 1/sqrt(8)
    #pragma unroll
    for (int hh = 0; hh < 4; ++hh) {
        float sc[KEIG];
        float mx = -3.0e38f;
        #pragma unroll
        for (int u = 0; u < KEIG; ++u) {
            float s = 0.f;
            #pragma unroll
            for (int d = 0; d < 8; ++d) s += q[hh * 8 + d] * ksh[u][hh * 8 + d];
            s *= scale;
            if (padsh[u]) s = -1e9f;
            sc[u] = s;
            mx = fmaxf(mx, s);
        }
        float sum = 0.f;
        #pragma unroll
        for (int u = 0; u < KEIG; ++u) { sc[u] = __expf(sc[u] - mx); sum += sc[u]; }
        const float inv = 1.f / sum;
        #pragma unroll
        for (int d = 0; d < 8; ++d) {
            float c = 0.f;
            #pragma unroll
            for (int u = 0; u < KEIG; ++u) c += sc[u] * vsh[u][hh * 8 + d];
            ctx[hh * 8 + d] = c * inv;
        }
    }
    // output projection
    float c2[32];
    #pragma unroll
    for (int j = 0; j < 32; ++j) {
        float s = bo[j];
        #pragma unroll
        for (int i = 0; i < 32; ++i) s += ctx[i] * Wo[j * 32 + i];
        c2[j] = s;
    }
    // filter MLP: tanh(relu(c2@Wf1^T+bf1)@Wf2^T+bf2)
    float acc = bf2[0];
    #pragma unroll
    for (int j = 0; j < 64; ++j) {
        float r = bf1[j];
        #pragma unroll
        for (int i = 0; i < 32; ++i) r += c2[i] * Wf1[j * 32 + i];
        if (r > 0.f) acc += r * Wf2[j];
    }
    const float f = tanhf(acc);
    filters[b * KEIG + t] = padsh[t] ? 0.f : f;
}

// ------------------ kernel 2: x_freq = V^T X  (per graph) -------------------
// grid = 64 blocks (one per graph), 256 threads = 8 waves.
// Per graph: M=32 (2 strips of 16), N=512 (32 tiles), K=1024 (32 steps of 32).
// Each wave owns 4 N-tiles but computes BOTH M-strips, so each x (B) fragment
// is loaded once and feeds two WMMAs: x traffic read exactly once per block.
__global__ __launch_bounds__(256) void xfreq_kernel(
    const float* __restrict__ x, const float* __restrict__ V,
    float* __restrict__ xfreq)
{
    const int b    = blockIdx.x;
    const int wave = threadIdx.x >> 5;
    const int lane = threadIdx.x & 31;
    const int hl   = lane >> 4;          // lane half
    const int mn   = lane & 15;          // M (for A) / N (for B,C)
    const int kh   = hl * 8;             // K-half base within 16-bit A/B layout

    const int ntb  = wave * 4;           // first of this wave's 4 n-tiles

    const float* Vb = V + (size_t)b * NPG * KEIG;
    const float* Xb = x + (size_t)b * NPG * DIN;

    v8f acc[2][4];
    #pragma unroll
    for (int s = 0; s < 2; ++s)
        #pragma unroll
        for (int t = 0; t < 4; ++t)
            acc[s][t] = (v8f){0.f,0.f,0.f,0.f,0.f,0.f,0.f,0.f};

    for (int ks = 0; ks < 32; ++ks) {
        const int kbase = ks * 32;
        // A[m][k] = V[node = kbase+k][eig]; strip 0: eig=mn, strip 1: eig=16+mn
        v16bf a0, a1;
        #pragma unroll
        for (int e = 0; e < 16; ++e) {
            const int k = (e < 8) ? (kh + e) : (16 + kh + (e - 8));
            const float* vrow = Vb + (size_t)(kbase + k) * KEIG;
            a0[e] = (__bf16)vrow[mn];
            a1[e] = (__bf16)vrow[16 + mn];
        }
        #pragma unroll
        for (int t = 0; t < 4; ++t) {
            const int n = (ntb + t) * 16 + mn;
            v16bf bm;
            #pragma unroll
            for (int e = 0; e < 16; ++e) {
                const int k = (e < 8) ? (kh + e) : (16 + kh + (e - 8));
                bm[e] = (__bf16)Xb[(size_t)(kbase + k) * DIN + n];
            }
            acc[0][t] = __builtin_amdgcn_wmma_f32_16x16x32_bf16(
                false, a0, false, bm, (short)0, acc[0][t], false, false);
            acc[1][t] = __builtin_amdgcn_wmma_f32_16x16x32_bf16(
                false, a1, false, bm, (short)0, acc[1][t], false, false);
        }
    }
    // C/D layout: vgpr r, lanes 0-15 -> M=r, lanes 16-31 -> M=8+r ; N = lane&15
    float* ob = xfreq + (size_t)b * KEIG * DIN;
    #pragma unroll
    for (int s = 0; s < 2; ++s) {
        #pragma unroll
        for (int t = 0; t < 4; ++t) {
            const int n = (ntb + t) * 16 + mn;
            #pragma unroll
            for (int r = 0; r < 8; ++r) {
                const int row = s * 16 + r + 8 * hl;
                ob[(size_t)row * DIN + n] = acc[s][t][r];
            }
        }
    }
}

// ------------------- Wp f32 -> bf16 (L2-resident operand) -------------------
__global__ __launch_bounds__(256) void cvt_wp_kernel(
    const float* __restrict__ Wp, __bf16* __restrict__ out, int n)
{
    const int i = blockIdx.x * 256 + threadIdx.x;
    if (i < n) out[i] = (__bf16)Wp[i];
}

// ------ kernel 3: fused  S=(V.diag(f))@x_freq ; out=LN(S@Wp^T + bp) ---------
// grid = 64 * 8 blocks (128-row tiles), 256 threads = 8 waves.
// LDS: S[128][512] bf16 (128KB) + O[128][512] bf16 (128KB) + stats (1KB)
#define ROWS 128
#define SMEM_BYTES (2 * ROWS * DOUT * 2 + 2 * ROWS * 4)

__global__ __launch_bounds__(256) void spectral_proj_kernel(
    const float* __restrict__ V, const float* __restrict__ xfreq,
    const float* __restrict__ filters, const __bf16* __restrict__ WpBf,
    const float* __restrict__ bp, const float* __restrict__ gp,
    const float* __restrict__ bep, float* __restrict__ out)
{
    extern __shared__ char smem[];
    __bf16* S     = (__bf16*)smem;                              // [128][512]
    __bf16* O     = (__bf16*)(smem + ROWS * DOUT * 2);          // [128][512]
    float*  rmean = (float*)(smem + 2 * ROWS * DOUT * 2);       // [128]
    float*  rrstd = rmean + ROWS;                               // [128]

    const int blk   = blockIdx.x;
    const int b     = blk >> 3;          // graph
    const int rbase = (blk & 7) * ROWS;  // row tile within graph

    const int wave = threadIdx.x >> 5;
    const int lane = threadIdx.x & 31;
    const int hl   = lane >> 4;
    const int mn   = lane & 15;
    const int kh   = hl * 8;

    const float* Vb = V + ((size_t)b * NPG + rbase) * KEIG;
    const float* Fb = xfreq + (size_t)b * KEIG * DIN;
    const float* fl = filters + b * KEIG;

    // ---- Stage A: S = (V . diag(f)) @ x_freq    (per wave: 16 rows, K=32)
    {
        const int m = wave * 16 + mn;    // row within the 128-row tile
        v16bf a;
        #pragma unroll
        for (int e = 0; e < 16; ++e) {
            const int k = (e < 8) ? (kh + e) : (16 + kh + (e - 8));
            a[e] = (__bf16)(Vb[(size_t)m * KEIG + k] * fl[k]);  // fold filters into V
        }
        for (int nt = 0; nt < 32; ++nt) {
            const int n = nt * 16 + mn;
            v16bf bm;
            #pragma unroll
            for (int e = 0; e < 16; ++e) {
                const int k = (e < 8) ? (kh + e) : (16 + kh + (e - 8));
                bm[e] = (__bf16)Fb[(size_t)k * DIN + n];
            }
            v8f c = (v8f){0.f,0.f,0.f,0.f,0.f,0.f,0.f,0.f};
            c = __builtin_amdgcn_wmma_f32_16x16x32_bf16(
                false, a, false, bm, (short)0, c, false, false);
            #pragma unroll
            for (int r = 0; r < 8; ++r) {
                const int row = wave * 16 + r + 8 * hl;
                S[row * DOUT + n] = (__bf16)c[r];
            }
        }
    }
    __syncthreads();

    // ---- Stage B: out_tile = S @ Wp^T + bp   (per wave: 16 rows x 512)
    const __bf16* Sw = S + wave * 16 * DIN;     // this wave's 16 A rows
    float rsum[8] = {0,0,0,0,0,0,0,0};
    float rsq [8] = {0,0,0,0,0,0,0,0};

    for (int nc = 0; nc < 4; ++nc) {            // 4 chunks of 128 output cols
        v8f acc[8];
        #pragma unroll
        for (int t = 0; t < 8; ++t) acc[t] = (v8f){0.f,0.f,0.f,0.f,0.f,0.f,0.f,0.f};

        for (int ks = 0; ks < 16; ++ks) {       // K = 512 in steps of 32
            // A from LDS: two contiguous 8xbf16 runs -> two ds_load_b128
            union { v16bf v; uint4 q[2]; } au;
            au.q[0] = *(const uint4*)(Sw + mn * DIN + ks * 32 + kh);
            au.q[1] = *(const uint4*)(Sw + mn * DIN + ks * 32 + 16 + kh);
            #pragma unroll
            for (int t = 0; t < 8; ++t) {
                const int n = nc * 128 + t * 16 + mn;  // dout
                const __bf16* wrow = WpBf + (size_t)n * DIN + ks * 32;
                union { v16bf v; uint4 q[2]; } bu;     // two global_load_b128
                bu.q[0] = *(const uint4*)(wrow + kh);
                bu.q[1] = *(const uint4*)(wrow + 16 + kh);
                // keep the L2-resident Wp stream warm in the near cache:
                // prefetch this row's fragment two K-steps ahead
                if (ks < 14) __builtin_prefetch(wrow + 64 + kh, 0, 1);
                acc[t] = __builtin_amdgcn_wmma_f32_16x16x32_bf16(
                    false, au.v, false, bu.v, (short)0, acc[t], false, false);
            }
        }
        // bias + LN stats accumulation + stage to LDS (bf16)
        #pragma unroll
        for (int t = 0; t < 8; ++t) {
            const int n = nc * 128 + t * 16 + mn;
            const float bias = bp[n];
            #pragma unroll
            for (int r = 0; r < 8; ++r) {
                const float v = acc[t][r] + bias;
                rsum[r] += v;
                rsq [r] += v * v;
                const int row = wave * 16 + r + 8 * hl;
                O[row * DOUT + n] = (__bf16)v;
            }
        }
    }

    // ---- per-row mean/var: butterfly over the 16 lanes of each half
    #pragma unroll
    for (int r = 0; r < 8; ++r) {
        float s = rsum[r], q = rsq[r];
        #pragma unroll
        for (int m = 8; m >= 1; m >>= 1) {
            s += __shfl_xor(s, m, 32);
            q += __shfl_xor(q, m, 32);
        }
        if (mn == 0) {
            const int row = wave * 16 + r + 8 * hl;
            const float mean = s * (1.f / (float)DOUT);
            const float var  = q * (1.f / (float)DOUT) - mean * mean;
            rmean[row] = mean;
            rrstd[row] = rsqrtf(var + 1e-5f);
        }
    }
    __syncthreads();

    // ---- normalize + streaming store
    float* ob = out + ((size_t)b * NPG + rbase) * DOUT;
    for (int idx = threadIdx.x; idx < ROWS * DOUT; idx += 256) {
        const int row = idx >> 9;
        const int col = idx & (DOUT - 1);
        const float v = (float)O[idx];
        ob[(size_t)row * DOUT + col] =
            (v - rmean[row]) * rrstd[row] * gp[col] + bep[col];
    }
}

// ---------------------------------------------------------------------------
extern "C" void kernel_launch(void* const* d_in, const int* in_sizes, int n_in,
                              void* d_out, int out_size, void* d_ws, size_t ws_size,
                              hipStream_t stream)
{
    const float* x    = (const float*)d_in[0];
    const float* eigv = (const float*)d_in[1];
    const float* eigs = (const float*)d_in[2];
    const unsigned char* mask = (const unsigned char*)d_in[3];
    // d_in[4] = batch (unused: equal-size graphs)
    const float* W1 = (const float*)d_in[5];
    const float* b1 = (const float*)d_in[6];
    const float* g1 = (const float*)d_in[7];
    const float* be1 = (const float*)d_in[8];
    const float* W2 = (const float*)d_in[9];
    const float* b2 = (const float*)d_in[10];
    const float* g2 = (const float*)d_in[11];
    const float* be2 = (const float*)d_in[12];
    const float* Wqkv = (const float*)d_in[13];
    const float* bqkv = (const float*)d_in[14];
    const float* Wo = (const float*)d_in[15];
    const float* bo = (const float*)d_in[16];
    const float* Wf1 = (const float*)d_in[17];
    const float* bf1 = (const float*)d_in[18];
    const float* Wf2 = (const float*)d_in[19];
    const float* bf2 = (const float*)d_in[20];
    const float* Wp = (const float*)d_in[21];
    const float* bp = (const float*)d_in[22];
    const float* gp = (const float*)d_in[23];
    const float* bep = (const float*)d_in[24];

    // workspace layout (all 16B aligned)
    float*  ws_filters = (float*)d_ws;                                   // 8 KB
    float*  ws_xfreq   = (float*)((char*)d_ws + 8192);                   // 4 MB
    __bf16* ws_wpbf    = (__bf16*)((char*)d_ws + 8192 +
                                   (size_t)NB * KEIG * DIN * 4);         // 512 KB

    // allow >64KB dynamic LDS (idempotent; capture-safe, not a stream op)
    static_assert(SMEM_BYTES <= 320 * 1024, "LDS over WGP budget");
    (void)hipFuncSetAttribute(reinterpret_cast<const void*>(spectral_proj_kernel),
                              hipFuncAttributeMaxDynamicSharedMemorySize, SMEM_BYTES);

    filters_kernel<<<dim3(NB), dim3(32), 0, stream>>>(
        eigs, mask, W1, b1, g1, be1, W2, b2, g2, be2,
        Wqkv, bqkv, Wo, bo, Wf1, bf1, Wf2, bf2, ws_filters);

    xfreq_kernel<<<dim3(NB), dim3(256), 0, stream>>>(x, eigv, ws_xfreq);

    cvt_wp_kernel<<<dim3((DOUT * DIN + 255) / 256), dim3(256), 0, stream>>>(
        Wp, ws_wpbf, DOUT * DIN);

    spectral_proj_kernel<<<dim3(NB * (NPG / ROWS)), dim3(256), SMEM_BYTES, stream>>>(
        eigv, ws_xfreq, ws_filters, ws_wpbf, bp, gp, bep, (float*)d_out);
}
